// MyKAMovement_73065983640225
// MI455X (gfx1250) — matrix operand order repaired and checked by reading, not compiled, not run on hardware
//
#include <hip/hip_runtime.h>
#include <hip/hip_bf16.h>

typedef __bf16 bf16;
typedef __attribute__((ext_vector_type(16))) __bf16 v16bf;
typedef __attribute__((ext_vector_type(8)))  float  v8f;

#define BB 16
#define PP 12
#define QQ 12
#define DV 64
#define NN 2000
#define CC 1024
#define KPAD 2016      /* 63*32, padded node dim for L GEMM K */
#define XHC 2048       /* 16 batches * 128 features */

// ---------------------------------------------------------------------------
// Generic bf16 WMMA GEMM: C(MxN) = act(A(MxK) @ B(KxN) + bias)
// A row-major (lda, lda%8==0, M>=1), B row-major (ldb, ldb%4==0). K must be
// a multiple of 32. B buffer must have ceil(N/64)*64 columns allocated
// (zero-padded). M,N arbitrary for the epilogue (guarded stores). A-tile rows
// beyond M are clamped to row M-1 (their results are never stored).
// Output f32 and/or bf16. act: 0=none 1=relu 2=sigmoid 3=tanh
// Tile 128x64, 256 threads = 8 waves (4x2), each wave 32x32 via 4 WMMAs.
// Double-buffered LDS: one barrier per K-step; global loads overlap WMMAs.
// ---------------------------------------------------------------------------
__global__ __launch_bounds__(256)
void k_gemm(const bf16* __restrict__ A, int lda,
            const bf16* __restrict__ B, int ldb,
            float* __restrict__ Cf, bf16* __restrict__ Cb, int ldc,
            const float* __restrict__ bias,
            int M, int N, int K, int act)
{
    __shared__ alignas(16) bf16 As[2][128][32];
    __shared__ alignas(16) bf16 Bs[2][64][32];

    const int tid  = threadIdx.x;
    const int wave = tid >> 5;
    const int lane = tid & 31;
    const int lr   = lane & 15;
    const int hi   = lane >> 4;
    const int m0   = blockIdx.y * 128;
    const int n0   = blockIdx.x * 64;
    const int wm   = (wave >> 1) * 32;   // 4 waves along M
    const int wn   = (wave & 1) * 32;    // 2 waves along N

    // A tile: 128 rows x 32 cols = 512 16B chunks; 2 per thread (rows clamped)
    const int ar0  = tid >> 2;             // rows 0..63
    const int ar1  = (tid >> 2) + 64;      // rows 64..127
    const int akc  = (tid & 3) * 8;        // k-chunk offset (elements)
    const int car0 = (m0 + ar0 < M) ? (m0 + ar0) : (M - 1);
    const int car1 = (m0 + ar1 < M) ? (m0 + ar1) : (M - 1);
    // B tile: 16 k-pairs x 16 col-groups-of-4; 1 per thread
    const int bkp  = tid >> 4;             // k = 2*bkp
    const int bnc  = (tid & 15) * 4;       // col group base

    v8f acc[2][2];
#pragma unroll
    for (int i = 0; i < 2; ++i)
#pragma unroll
        for (int j = 0; j < 2; ++j)
#pragma unroll
            for (int t = 0; t < 8; ++t) acc[i][j][t] = 0.0f;

    uint4 ra0, ra1;
    uint2 rb0, rb1;

    auto load_tiles = [&](int k0) {
        ra0 = *(const uint4*)(A + (long)car0 * lda + k0 + akc);
        ra1 = *(const uint4*)(A + (long)car1 * lda + k0 + akc);
        const bf16* bp = B + (long)(k0 + 2 * bkp) * ldb + n0 + bnc;
        rb0 = *(const uint2*)bp;
        rb1 = *(const uint2*)(bp + ldb);
    };
    auto store_tiles = [&](int buf) {
        *(uint4*)(&As[buf][ar0][akc]) = ra0;
        *(uint4*)(&As[buf][ar1][akc]) = ra1;
        union { uint2 v; unsigned short s[4]; } u0, u1;
        u0.v = rb0; u1.v = rb1;
#pragma unroll
        for (int i = 0; i < 4; ++i)
            *(unsigned*)(&Bs[buf][bnc + i][2 * bkp]) =
                (unsigned)u0.s[i] | ((unsigned)u1.s[i] << 16);
    };

    load_tiles(0);
    store_tiles(0);

    int buf = 0;
    for (int k0 = 0; k0 < K; k0 += 32, buf ^= 1) {
        __syncthreads();   // LDS[buf] published by all waves
        const bool more = (k0 + 32 < K);
        if (more) {
            load_tiles(k0 + 32);   // global loads overlap WMMAs below
            if (k0 + 64 < K)
                __builtin_prefetch(B + (long)(k0 + 64 + 2 * bkp) * ldb + n0 + bnc, 0, 1);
        }

        union Frag { v16bf v; unsigned u[8]; } fa[2], fb[2];
#pragma unroll
        for (int i = 0; i < 2; ++i) {
            const unsigned* ap = (const unsigned*)(&As[buf][wm + i * 16 + lr][0]);
#pragma unroll
            for (int j = 0; j < 4; ++j) {
                fa[i].u[j]     = ap[hi * 4 + j];       // k = hi*8 + 2j..2j+1
                fa[i].u[4 + j] = ap[8 + hi * 4 + j];   // k = 16 + hi*8 + 2j..
            }
        }
#pragma unroll
        for (int j = 0; j < 2; ++j) {
            const unsigned* bp = (const unsigned*)(&Bs[buf][wn + j * 16 + lr][0]);
#pragma unroll
            for (int t = 0; t < 8; ++t) fb[j].u[t] = bp[(t >> 2) * 8 + hi * 4 + (t & 3)]; // k = (t>>2)*16 + hi*8 + 2(t&3)..
        }
#pragma unroll
        for (int i = 0; i < 2; ++i)
#pragma unroll
            for (int j = 0; j < 2; ++j)
                acc[i][j] = __builtin_amdgcn_wmma_f32_16x16x32_bf16(
                    false, fa[i].v, false, fb[j].v, (short)0, acc[i][j], false, false);

        if (more) store_tiles(buf ^ 1);   // other waves may still read LDS[buf]
    }

#pragma unroll
    for (int i = 0; i < 2; ++i)
#pragma unroll
        for (int j = 0; j < 2; ++j)
#pragma unroll
            for (int t = 0; t < 8; ++t) {
                int row = m0 + wm + i * 16 + hi * 8 + t;
                int col = n0 + wn + j * 16 + lr;
                if (row < M && col < N) {
                    float v = acc[i][j][t];
                    if (bias) v += bias[col];
                    if (act == 1)      v = fmaxf(v, 0.0f);
                    else if (act == 2) v = 1.0f / (1.0f + __expf(-v));
                    else if (act == 3) v = tanhf(v);
                    long o = (long)row * ldc + col;
                    if (Cf) Cf[o] = v;
                    if (Cb) Cb[o] = (bf16)v;
                }
            }
}

// ---------------------------------------------------------------------------
// Small glue kernels
// ---------------------------------------------------------------------------
__global__ void k_cast_bf16(const float* __restrict__ s, bf16* __restrict__ d, long n)
{
    long i = (long)blockIdx.x * blockDim.x + threadIdx.x;
    if (i < n) d[i] = (bf16)s[i];
}

// cast (rows x cols) f32 into (rows x ldd) bf16 with zero column padding
__global__ void k_cast_pad(const float* __restrict__ s, bf16* __restrict__ d,
                           int rows, int cols, int ldd)
{
    int i = blockIdx.x * blockDim.x + threadIdx.x;
    if (i >= rows * ldd) return;
    int r = i / ldd, c = i % ldd;
    d[i] = (c < cols) ? (bf16)s[r * cols + c] : (bf16)0.0f;
}

// relu(x@w1+b1)@w2+b2 over rows of 64 features (SE embedding)
__global__ void k_mlp2_64(const float* __restrict__ in,
                          const float* __restrict__ w1, const float* __restrict__ b1,
                          const float* __restrict__ w2, const float* __restrict__ b2,
                          float* __restrict__ out)
{
    __shared__ float h[64];
    int r = blockIdx.x, t = threadIdx.x;
    const float* x = in + (long)r * 64;
    float s = b1[t];
    for (int k = 0; k < 64; ++k) s += x[k] * w1[k * 64 + t];
    h[t] = fmaxf(s, 0.0f);
    __syncthreads();
    float o = b2[t];
    for (int k = 0; k < 64; ++k) o += h[k] * w2[k * 64 + t];
    out[(long)r * 64 + t] = o;
}

// TE one-hot MLP: gather tw1 rows instead of materializing one-hot
__global__ void k_te_embed(const int* __restrict__ TE,
                           const float* __restrict__ w1, const float* __restrict__ b1,
                           const float* __restrict__ w2, const float* __restrict__ b2,
                           float* __restrict__ out)
{
    __shared__ float h[64];
    int r = blockIdx.x;                 // b*PP + p
    int b = r / PP, p = r % PP;
    int t = threadIdx.x;
    int te0 = TE[((long)b * (PP + QQ) + p) * 2 + 0];
    int te1 = TE[((long)b * (PP + QQ) + p) * 2 + 1];
    float s = w1[te0 * 64 + t] + w1[(7 + te1) * 64 + t] + b1[t];
    h[t] = fmaxf(s, 0.0f);
    __syncthreads();
    float o = b2[t];
    for (int k = 0; k < 64; ++k) o += h[k] * w2[k * 64 + t];
    out[(long)r * 64 + t] = o;
}

// Zf = mlp2(Z) + STEZ  -> bf16 (B*P*C, 64)
__global__ void k_zf(const float* __restrict__ Z,
                     const float* __restrict__ w1, const float* __restrict__ b1,
                     const float* __restrict__ w2, const float* __restrict__ b2,
                     const float* __restrict__ sez, const float* __restrict__ tez,
                     bf16* __restrict__ Zf)
{
    __shared__ float h[64];
    int r = blockIdx.x;                 // (b*PP+p)*1024 + cell
    int c = r & 1023, bp = r >> 10;
    int t = threadIdx.x;
    float z0 = Z[(long)r * 2 + 0], z1 = Z[(long)r * 2 + 1];
    h[t] = fmaxf(z0 * w1[t] + z1 * w1[64 + t] + b1[t], 0.0f);
    __syncthreads();
    float o = b2[t];
    for (int k = 0; k < 64; ++k) o += h[k] * w2[k * 64 + t];
    o += sez[(long)c * 64 + t] + tez[(long)bp * 64 + t];
    Zf[(long)r * 64 + t] = (bf16)o;
}

// im2col for ConvLSTM: col(16384, 1152) = [patches(x) | patches(h_prev)]
__global__ void k_im2col(const bf16* __restrict__ Zf, const float* __restrict__ hcl,
                         bf16* __restrict__ col, int p)
{
    long i = (long)blockIdx.x * blockDim.x + threadIdx.x;
    if (i >= (long)16384 * 1152) return;
    int row = (int)(i / 1152);
    int cc  = (int)(i % 1152);
    int half = cc / 576;
    int rem  = cc % 576;
    int tap = rem / DV, ci = rem % DV;
    int b = row >> 10, pix = row & 1023;
    int y = pix >> 5, x = pix & 31;
    int yy = y + tap / 3 - 1, xx = x + tap % 3 - 1;
    bf16 v = (bf16)0.0f;
    if (yy >= 0 && yy < 32 && xx >= 0 && xx < 32) {
        int spix = yy * 32 + xx;
        if (half == 0) v = Zf[((long)((b * PP + p) * 1024 + spix)) * DV + ci];
        else           v = (bf16)hcl[((long)(b * 1024 + spix)) * DV + ci];
    }
    col[i] = v;
}

// ConvLSTM pointwise update; scatter h into cell-major ZcPerm(1024, B*P*64)
__global__ void k_convlstm_pw(const float* __restrict__ gates,
                              float* __restrict__ ccl, float* __restrict__ hcl,
                              bf16* __restrict__ ZcPerm, int p)
{
    long i = (long)blockIdx.x * blockDim.x + threadIdx.x;
    if (i >= (long)16384 * DV) return;
    int row = (int)(i >> 6), d = (int)(i & 63);
    const float* g = gates + (long)row * 256;
    float ig = g[d], fg = g[64 + d], gg = g[128 + d], og = g[192 + d];
    auto hs = [](float x) { return fminf(fmaxf(0.2f * x + 0.5f, 0.0f), 1.0f); };
    float c = hs(fg) * ccl[i] + hs(ig) * tanhf(gg);
    float h = hs(og) * tanhf(c);
    ccl[i] = c;
    hcl[i] = h;
    int b = row >> 10, pix = row & 1023;
    ZcPerm[(long)pix * (BB * PP * DV) + (b * PP + p) * DV + d] = (bf16)h;
}

// gumbel softmax over rows of (C,C) -> bf16
__global__ void k_gumbel_softmax(const float* __restrict__ logits,
                                 const float* __restrict__ noise,
                                 bf16* __restrict__ out)
{
    __shared__ float red[256];
    int r = blockIdx.x, t = threadIdx.x;
    float mx = -1e30f;
    for (int j = t; j < CC; j += 256) {
        float u = noise[(long)r * CC + j];
        float g = -logf(-logf(u + 1e-20f) + 1e-20f);
        mx = fmaxf(mx, logits[(long)r * CC + j] + g);
    }
    red[t] = mx; __syncthreads();
    for (int s = 128; s > 0; s >>= 1) { if (t < s) red[t] = fmaxf(red[t], red[t + s]); __syncthreads(); }
    mx = red[0]; __syncthreads();
    float sum = 0.0f;
    for (int j = t; j < CC; j += 256) {
        float u = noise[(long)r * CC + j];
        float g = -logf(-logf(u + 1e-20f) + 1e-20f);
        sum += __expf(logits[(long)r * CC + j] + g - mx);
    }
    red[t] = sum; __syncthreads();
    for (int s = 128; s > 0; s >>= 1) { if (t < s) red[t] += red[t + s]; __syncthreads(); }
    float inv = 1.0f / red[0];
    for (int j = t; j < CC; j += 256) {
        float u = noise[(long)r * CC + j];
        float g = -logf(-logf(u + 1e-20f) + 1e-20f);
        out[(long)r * CC + j] = (bf16)(__expf(logits[(long)r * CC + j] + g - mx) * inv);
    }
}

// scaled laplacian: d[n] = sum_m max(A[n,m],A[m,n])
__global__ void k_lap_rows(const float* __restrict__ adj, float* __restrict__ d)
{
    __shared__ float red[256];
    int n = blockIdx.x, t = threadIdx.x;
    float s = 0.0f;
    for (int m = t; m < NN; m += 256) s += fmaxf(adj[(long)n * NN + m], adj[(long)m * NN + n]);
    red[t] = s; __syncthreads();
    for (int k = 128; k > 0; k >>= 1) { if (t < k) red[t] += red[t + k]; __syncthreads(); }
    if (t == 0) d[n] = red[0];
}

__global__ void k_dinv(const float* __restrict__ d, float* __restrict__ dinv)
{
    int n = blockIdx.x * blockDim.x + threadIdx.x;
    if (n < NN) dinv[n] = d[n] > 0.0f ? rsqrtf(d[n]) : 0.0f;
}

// L_tilde = -(dinv_n * max(A,At)_nm * dinv_m), zero-padded cols to KPAD
__global__ void k_lap_fill(const float* __restrict__ adj, const float* __restrict__ dinv,
                           bf16* __restrict__ Lb)
{
    long i = (long)blockIdx.x * blockDim.x + threadIdx.x;
    if (i >= (long)NN * KPAD) return;
    int n = (int)(i / KPAD), m = (int)(i % KPAD);
    float v = 0.0f;
    if (m < NN) v = -dinv[n] * dinv[m] * fmaxf(adj[(long)n * NN + m], adj[(long)m * NN + n]);
    Lb[i] = (bf16)v;
}

// Xf = mlp2(X[...,None]) + STEX + ZX  -> bf16 ((p*N+n)*16+b, 64)
__global__ void k_xf(const float* __restrict__ X,
                     const float* __restrict__ w1, const float* __restrict__ b1,
                     const float* __restrict__ w2, const float* __restrict__ b2,
                     const float* __restrict__ sex, const float* __restrict__ tex,
                     const bf16* __restrict__ ZXbf,
                     bf16* __restrict__ Xf_all)
{
    __shared__ float h[64];
    int r = blockIdx.x;                 // (p*NN + n)*16 + b
    int b = r & 15, pn = r >> 4;
    int n = pn % NN, p = pn / NN;
    int t = threadIdx.x;
    float x = X[((long)b * PP + p) * NN + n];
    h[t] = fmaxf(x * w1[t] + b1[t], 0.0f);
    __syncthreads();
    float o = b2[t];
    for (int k = 0; k < 64; ++k) o += h[k] * w2[k * 64 + t];
    o += sex[(long)n * 64 + t] + tex[((long)b * PP + p) * 64 + t];
    o += (float)ZXbf[(long)n * (BB * PP * DV) + (b * PP + p) * DV + t];
    Xf_all[(long)r * 64 + t] = (bf16)o;
}

// pack xh(n, b*128+f) = [Xf | (r?)*h]
__global__ void k_pack_xh(const bf16* __restrict__ Xf_p, const float* __restrict__ h,
                          const float* __restrict__ ru, int use_r,
                          bf16* __restrict__ xh)
{
    long i = (long)blockIdx.x * blockDim.x + threadIdx.x;
    if (i >= (long)NN * XHC) return;
    int f = (int)(i & 127);
    int nb = (int)(i >> 7);
    int b = nb & 15, n = nb >> 4;
    bf16 v;
    if (f < 64) v = Xf_p[((long)n * 16 + b) * 64 + f];
    else {
        float hv = h[((long)n * 16 + b) * 64 + (f - 64)];
        if (use_r) hv *= ru[((long)n * 16 + b) * 128 + (f - 64)];
        v = (bf16)hv;
    }
    xh[(long)n * XHC + b * 128 + f] = v;
}

// Chebyshev combine: out = bf16(2*T - prev)   (rows < NN only; pads stay 0)
__global__ void k_cheb(const float* __restrict__ T, const bf16* __restrict__ prev,
                       bf16* __restrict__ out)
{
    long i = (long)blockIdx.x * blockDim.x + threadIdx.x;
    if (i >= (long)NN * XHC) return;
    out[i] = (bf16)(2.0f * T[i] - (float)prev[i]);
}

// G[(n*16+b), f*4+k] = xs_k[n, b*128+f]  (matches stack(...,-1).reshape order)
__global__ void k_pack_G(const bf16* __restrict__ x0, const bf16* __restrict__ x1,
                         const bf16* __restrict__ x2, const bf16* __restrict__ x3,
                         bf16* __restrict__ G)
{
    long i = (long)blockIdx.x * blockDim.x + threadIdx.x;
    if (i >= (long)32000 * 512) return;
    int c = (int)(i & 511), row = (int)(i >> 9);
    int kt = c & 3, f = c >> 2;
    int b = row & 15, n = row >> 4;
    const bf16* src = (kt == 0) ? x0 : (kt == 1) ? x1 : (kt == 2) ? x2 : x3;
    G[i] = src[(long)n * XHC + b * 128 + f];
}

__global__ void k_gru_update(const float* __restrict__ ru, const float* __restrict__ cnd,
                             float* __restrict__ h)
{
    long i = (long)blockIdx.x * blockDim.x + threadIdx.x;
    if (i >= (long)32000 * 64) return;
    int row = (int)(i >> 6), f = (int)(i & 63);
    float u = ru[(long)row * 128 + 64 + f];
    h[i] = u * h[i] + (1.0f - u) * cnd[i];
}

__global__ void k_out_tr(const float* __restrict__ Y, float* __restrict__ out)
{
    int i = blockIdx.x * blockDim.x + threadIdx.x;
    if (i >= BB * QQ * NN) return;
    int n = i % NN, q = (i / NN) % QQ, b = i / (NN * QQ);
    out[i] = Y[((long)n * 16 + b) * QQ + q];
}

// ---------------------------------------------------------------------------
extern "C" void kernel_launch(void* const* d_in, const int* in_sizes, int n_in,
                              void* d_out, int out_size, void* d_ws, size_t ws_size,
                              hipStream_t stream)
{
    const float* X          = (const float*)d_in[0];
    const float* Z          = (const float*)d_in[1];
    const int*   TE         = (const int*)  d_in[2];
    const float* adj_gg     = (const float*)d_in[3];
    const float* adj_gr     = (const float*)d_in[4];
    const float* gnoise     = (const float*)d_in[5];
    const float* se_x       = (const float*)d_in[6];
    const float* se_z       = (const float*)d_in[7];
    const float* mov_logits = (const float*)d_in[8];
    const float* stex_se_w1 = (const float*)d_in[9];
    const float* stex_se_b1 = (const float*)d_in[10];
    const float* stex_se_w2 = (const float*)d_in[11];
    const float* stex_se_b2 = (const float*)d_in[12];
    const float* stex_te_w1 = (const float*)d_in[13];
    const float* stex_te_b1 = (const float*)d_in[14];
    const float* stex_te_w2 = (const float*)d_in[15];
    const float* stex_te_b2 = (const float*)d_in[16];
    const float* stez_se_w1 = (const float*)d_in[17];
    const float* stez_se_b1 = (const float*)d_in[18];
    const float* stez_se_w2 = (const float*)d_in[19];
    const float* stez_se_b2 = (const float*)d_in[20];
    const float* stez_te_w1 = (const float*)d_in[21];
    const float* stez_te_b1 = (const float*)d_in[22];
    const float* stez_te_w2 = (const float*)d_in[23];
    const float* stez_te_b2 = (const float*)d_in[24];
    const float* fcs_z_w1   = (const float*)d_in[25];
    const float* fcs_z_b1   = (const float*)d_in[26];
    const float* fcs_z_w2   = (const float*)d_in[27];
    const float* fcs_z_b2   = (const float*)d_in[28];
    const float* fcs_x_in_w1= (const float*)d_in[29];
    const float* fcs_x_in_b1= (const float*)d_in[30];
    const float* fcs_x_in_w2= (const float*)d_in[31];
    const float* fcs_x_in_b2= (const float*)d_in[32];
    const float* mov_fc_w   = (const float*)d_in[33];
    const float* mov_fc_b   = (const float*)d_in[34];
    const float* out_w1     = (const float*)d_in[35];
    const float* out_b1     = (const float*)d_in[36];
    const float* out_w2     = (const float*)d_in[37];
    const float* out_b2     = (const float*)d_in[38];
    const float* conv_wx    = (const float*)d_in[39];
    const float* conv_wh    = (const float*)d_in[40];
    const float* gate_w     = (const float*)d_in[41];
    const float* gate_b     = (const float*)d_in[42];
    const float* cand_w     = (const float*)d_in[43];
    const float* cand_b     = (const float*)d_in[44];
    (void)in_sizes; (void)n_in; (void)out_size; (void)ws_size;

    char* base = (char*)d_ws;
    size_t off = 0;
    auto alloc = [&](size_t bytes) -> char* {
        char* p = base + off;
        off = (off + bytes + 255) & ~(size_t)255;
        return p;
    };

    // persistent buffers
    bf16*  adjgr_bf = (bf16*) alloc((size_t)NN * CC * 2);
    bf16*  gateW    = (bf16*) alloc(512 * 128 * 2);
    bf16*  candW    = (bf16*) alloc(512 * 64 * 2);
    bf16*  movW     = (bf16*) alloc(64 * 64 * 2);
    bf16*  outW1    = (bf16*) alloc(64 * 64 * 2);
    bf16*  outW2    = (bf16*) alloc(64 * 64 * 2);   // (64x12) padded to 64 cols
    bf16*  convW    = (bf16*) alloc((size_t)1152 * 256 * 2);
    bf16*  mZ       = (bf16*) alloc((size_t)CC * CC * 2);
    float* sex      = (float*)alloc((size_t)NN * 64 * 4);
    float* sez      = (float*)alloc((size_t)CC * 64 * 4);
    float* tex      = (float*)alloc((size_t)BB * PP * 64 * 4);
    float* tez      = (float*)alloc((size_t)BB * PP * 64 * 4);
    float* dsum     = (float*)alloc(NN * 4);
    float* dinv     = (float*)alloc(NN * 4);
    bf16*  Lb       = (bf16*) alloc((size_t)NN * KPAD * 2);
    float* h_gru    = (float*)alloc((size_t)32000 * 64 * 4);
    bf16*  Xf_all   = (bf16*) alloc((size_t)PP * 32000 * 64 * 2);
    bf16*  Zf       = (bf16*) alloc((size_t)196608 * 64 * 2);
    bf16*  ZcPerm   = (bf16*) alloc((size_t)CC * 12288 * 2);
    bf16*  ZXbf     = (bf16*) alloc((size_t)NN * 12288 * 2);
    // reusable scratch arena
    char*  S        = alloc((size_t)120 << 20);
    size_t soff = 0;
    auto salloc = [&](size_t bytes) -> char* {
        char* p = S + soff;
        soff = (soff + bytes + 255) & ~(size_t)255;
        return p;
    };

    auto gemm = [&](const bf16* A, int lda, const bf16* Bm, int ldb,
                    float* Cf, bf16* Cb, int ldc, const float* bias,
                    int M, int N, int K, int act) {
        dim3 g((N + 63) / 64, (M + 127) / 128);
        k_gemm<<<g, 256, 0, stream>>>(A, lda, Bm, ldb, Cf, Cb, ldc, bias, M, N, K, act);
    };
    auto cast = [&](const float* s, bf16* d, long n) {
        k_cast_bf16<<<(int)((n + 255) / 256), 256, 0, stream>>>(s, d, n);
    };

    // ---- weight casts ----
    cast(adj_gr, adjgr_bf, (long)NN * CC);
    cast(gate_w, gateW, 512L * 128);
    cast(cand_w, candW, 512L * 64);
    cast(mov_fc_w, movW, 64L * 64);
    cast(out_w1, outW1, 64L * 64);
    k_cast_pad<<<(64 * 64 + 255) / 256, 256, 0, stream>>>(out_w2, outW2, 64, 12, 64);
    cast(conv_wx, convW, 576L * 256);
    cast(conv_wh, convW + 576L * 256, 576L * 256);

    // ---- embeddings ----
    k_mlp2_64<<<NN, 64, 0, stream>>>(se_x, stex_se_w1, stex_se_b1, stex_se_w2, stex_se_b2, sex);
    k_mlp2_64<<<CC, 64, 0, stream>>>(se_z, stez_se_w1, stez_se_b1, stez_se_w2, stez_se_b2, sez);
    k_te_embed<<<BB * PP, 64, 0, stream>>>(TE, stex_te_w1, stex_te_b1, stex_te_w2, stex_te_b2, tex);
    k_te_embed<<<BB * PP, 64, 0, stream>>>(TE, stez_te_w1, stez_te_b1, stez_te_w2, stez_te_b2, tez);

    // ---- cell branch: Zf then ConvLSTM (im2col + WMMA GEMM) ----
    k_zf<<<196608, 64, 0, stream>>>(Z, fcs_z_w1, fcs_z_b1, fcs_z_w2, fcs_z_b2, sez, tez, Zf);

    soff = 0;
    float* h_cl  = (float*)salloc((size_t)16384 * 64 * 4);
    float* c_cl  = (float*)salloc((size_t)16384 * 64 * 4);
    bf16*  col   = (bf16*) salloc((size_t)16384 * 1152 * 2);
    float* gates = (float*)salloc((size_t)16384 * 256 * 4);
    hipMemsetAsync(h_cl, 0, (size_t)16384 * 64 * 4, stream);
    hipMemsetAsync(c_cl, 0, (size_t)16384 * 64 * 4, stream);
    {
        long ic = 16384L * 1152, pw = 16384L * 64;
        for (int p = 0; p < PP; ++p) {
            k_im2col<<<(int)((ic + 255) / 256), 256, 0, stream>>>(Zf, h_cl, col, p);
            gemm(col, 1152, convW, 256, gates, nullptr, 256, nullptr, 16384, 256, 1152, 0);
            k_convlstm_pw<<<(int)((pw + 255) / 256), 256, 0, stream>>>(gates, c_cl, h_cl, ZcPerm, p);
        }
    }

    // ---- gumbel routing + adj_gr projection ----
    k_gumbel_softmax<<<CC, 256, 0, stream>>>(mov_logits, gnoise, mZ);
    soff = 0;
    bf16* Zc2 = (bf16*)salloc((size_t)CC * 12288 * 2);
    bf16* Zc3 = (bf16*)salloc((size_t)CC * 12288 * 2);
    gemm(mZ, CC, ZcPerm, 12288, nullptr, Zc2, 12288, nullptr, CC, 12288, CC, 0);
    gemm(Zc2, 64, movW, 64, nullptr, Zc3, 64, mov_fc_b, 196608, 64, 64, 1);
    gemm(adjgr_bf, CC, Zc3, 12288, nullptr, ZXbf, 12288, nullptr, NN, 12288, CC, 0);

    // ---- laplacian ----
    k_lap_rows<<<NN, 256, 0, stream>>>(adj_gg, dsum);
    k_dinv<<<(NN + 255) / 256, 256, 0, stream>>>(dsum, dinv);
    {
        long lt = (long)NN * KPAD;
        k_lap_fill<<<(int)((lt + 255) / 256), 256, 0, stream>>>(adj_gg, dinv, Lb);
    }

    // ---- node-branch input features ----
    k_xf<<<PP * NN * BB, 64, 0, stream>>>(X, fcs_x_in_w1, fcs_x_in_b1, fcs_x_in_w2, fcs_x_in_b2,
                                          sex, tex, ZXbf, Xf_all);

    // ---- DCGRU ----
    soff = 0;
    bf16*  xh0 = (bf16*) salloc((size_t)KPAD * XHC * 2);
    bf16*  x1  = (bf16*) salloc((size_t)KPAD * XHC * 2);
    bf16*  x2  = (bf16*) salloc((size_t)KPAD * XHC * 2);
    bf16*  x3  = (bf16*) salloc((size_t)KPAD * XHC * 2);
    float* T   = (float*)salloc((size_t)NN * XHC * 4);
    bf16*  G   = (bf16*) salloc((size_t)32000 * 512 * 2);
    float* ru  = (float*)salloc((size_t)32000 * 128 * 4);
    float* cnd = (float*)salloc((size_t)32000 * 64 * 4);
    hipMemsetAsync(h_gru, 0, (size_t)32000 * 64 * 4, stream);
    hipMemsetAsync(xh0, 0, (size_t)KPAD * XHC * 2, stream);
    hipMemsetAsync(x1,  0, (size_t)KPAD * XHC * 2, stream);
    hipMemsetAsync(x2,  0, (size_t)KPAD * XHC * 2, stream);
    hipMemsetAsync(x3,  0, (size_t)KPAD * XHC * 2, stream);

    long xt = (long)NN * XHC, gt = 32000L * 512, ht = 32000L * 64;
    for (int p = 0; p < PP; ++p) {
        const bf16* Xf_p = Xf_all + (size_t)p * 32000 * 64;
        // gate gconv: xs from concat(x, h)
        k_pack_xh<<<(int)((xt + 255) / 256), 256, 0, stream>>>(Xf_p, h_gru, nullptr, 0, xh0);
        gemm(Lb, KPAD, xh0, XHC, nullptr, x1, XHC, nullptr, NN, XHC, KPAD, 0);
        gemm(Lb, KPAD, x1, XHC, T, nullptr, XHC, nullptr, NN, XHC, KPAD, 0);
        k_cheb<<<(int)((xt + 255) / 256), 256, 0, stream>>>(T, xh0, x2);
        gemm(Lb, KPAD, x2, XHC, T, nullptr, XHC, nullptr, NN, XHC, KPAD, 0);
        k_cheb<<<(int)((xt + 255) / 256), 256, 0, stream>>>(T, x1, x3);
        k_pack_G<<<(int)((gt + 255) / 256), 256, 0, stream>>>(xh0, x1, x2, x3, G);
        gemm(G, 512, gateW, 128, ru, nullptr, 128, gate_b, 32000, 128, 512, 2);
        // candidate gconv: xs from concat(x, r*h)
        k_pack_xh<<<(int)((xt + 255) / 256), 256, 0, stream>>>(Xf_p, h_gru, ru, 1, xh0);
        gemm(Lb, KPAD, xh0, XHC, nullptr, x1, XHC, nullptr, NN, XHC, KPAD, 0);
        gemm(Lb, KPAD, x1, XHC, T, nullptr, XHC, nullptr, NN, XHC, KPAD, 0);
        k_cheb<<<(int)((xt + 255) / 256), 256, 0, stream>>>(T, xh0, x2);
        gemm(Lb, KPAD, x2, XHC, T, nullptr, XHC, nullptr, NN, XHC, KPAD, 0);
        k_cheb<<<(int)((xt + 255) / 256), 256, 0, stream>>>(T, x1, x3);
        k_pack_G<<<(int)((gt + 255) / 256), 256, 0, stream>>>(xh0, x1, x2, x3, G);
        gemm(G, 512, candW, 64, cnd, nullptr, 64, cand_b, 32000, 64, 512, 3);
        k_gru_update<<<(int)((ht + 255) / 256), 256, 0, stream>>>(ru, cnd, h_gru);
    }

    // ---- output MLP + transpose ----
    soff = 0;
    bf16*  h_bf = (bf16*) salloc((size_t)32000 * 64 * 2);
    bf16*  hid  = (bf16*) salloc((size_t)32000 * 64 * 2);
    float* Y    = (float*)salloc((size_t)32000 * 12 * 4);
    cast(h_gru, h_bf, 32000L * 64);
    gemm(h_bf, 64, outW1, 64, nullptr, hid, 64, out_b1, 32000, 64, 64, 1);
    gemm(hid, 64, outW2, 64, Y, nullptr, 12, out_b2, 32000, 12, 64, 0);
    {
        int ot = BB * QQ * NN;
        k_out_tr<<<(ot + 255) / 256, 256, 0, stream>>>(Y, (float*)d_out);
    }
}
